// MultiHeadAttention_27384711480121
// MI455X (gfx1250) — matrix operand order repaired and checked by reading, not compiled, not run on hardware
//
#include <hip/hip_runtime.h>
#include <hip/hip_bf16.h>
#include <math.h>

#define EMBED 1024
#define NHEAD 16
#define HD    64
#define BB    4
#define LL    2048

typedef __attribute__((ext_vector_type(16))) _Float16 v16h;
typedef __attribute__((ext_vector_type(8)))  float    v8f;

union AFrag16 { v16h v; _Float16 h[16]; uint4 q[2]; };

__device__ __forceinline__ v8f wmma_f16(v16h a, v16h b, v8f c) {
  // v_wmma_f32_16x16x32_f16: (neg_a, A, neg_b, B, c_mod, C, reuse_a, reuse_b)
  return __builtin_amdgcn_wmma_f32_16x16x32_f16(false, a, false, b, (short)0, c, false, false);
}

// ---------------------------------------------------------------------------
// Kernel 1: qkv = x @ w_qkv + b_qkv  (M=8192, N=3072, K=1024)
// Scatter epilogue -> Q (f16, pre-scaled by 1/sqrt(64)), K (f16, [b,h,l,d]),
// V transposed (f16, [b,h,d,l]).
// ---------------------------------------------------------------------------
__global__ __launch_bounds__(256)
void qkv_kernel(const float* __restrict__ x, const float* __restrict__ w,
                const float* __restrict__ bias,
                _Float16* __restrict__ Qb, _Float16* __restrict__ Kb,
                _Float16* __restrict__ Vt)
{
  const int N = 3 * EMBED;      // 3072
  const int K = EMBED;          // 1024
  __shared__ _Float16 As[128][40];  // [M][K] tile, pad to 40 halfs (16B aligned)
  __shared__ _Float16 Bs[64][40];   // [N][K] tile (transposed on fill)

  const int t    = threadIdx.x;
  const int lane = t & 31;
  const int wid  = t >> 5;          // 0..7
  const int wm   = wid >> 1;        // 0..3  (32-row strip)
  const int wn   = wid & 1;         // 0..1  (32-col strip)
  const int half = lane >> 4;
  const int l16  = lane & 15;
  const int m0   = blockIdx.x * 128;
  const int n0   = blockIdx.y * 64;

  v8f acc[2][2];
  for (int i = 0; i < 2; ++i) for (int j = 0; j < 2; ++j) acc[i][j] = {};

  for (int k0 = 0; k0 < K; k0 += 32) {
    // A tile: 128x32 f32 -> f16
    for (int it = 0; it < 4; ++it) {
      int idx = t + it * 256;                 // 0..1023
      int row = idx >> 3;
      int c4  = (idx & 7) << 2;
      float4 f = *(const float4*)(x + (size_t)(m0 + row) * K + k0 + c4);
      As[row][c4 + 0] = (_Float16)f.x;  As[row][c4 + 1] = (_Float16)f.y;
      As[row][c4 + 2] = (_Float16)f.z;  As[row][c4 + 3] = (_Float16)f.w;
    }
    // B tile: 32x64 f32 -> f16, transposed into [n][k]
    for (int it = 0; it < 2; ++it) {
      int idx = t + it * 256;                 // 0..511
      int kr  = idx >> 4;
      int c4  = (idx & 15) << 2;
      float4 f = *(const float4*)(w + (size_t)(k0 + kr) * N + n0 + c4);
      Bs[c4 + 0][kr] = (_Float16)f.x;  Bs[c4 + 1][kr] = (_Float16)f.y;
      Bs[c4 + 2][kr] = (_Float16)f.z;  Bs[c4 + 3][kr] = (_Float16)f.w;
    }
    __syncthreads();

    AFrag16 a[2], b[2];
    for (int i = 0; i < 2; ++i) {
      const _Float16* p = &As[wm * 32 + i * 16 + l16][half * 8];
      a[i].q[0] = *(const uint4*)p;
      a[i].q[1] = *(const uint4*)(p + 16);
    }
    for (int j = 0; j < 2; ++j) {
      const _Float16* p = &Bs[wn * 32 + j * 16 + l16][half * 8];
      b[j].q[0] = *(const uint4*)p;
      b[j].q[1] = *(const uint4*)(p + 16);
    }
    for (int i = 0; i < 2; ++i)
      for (int j = 0; j < 2; ++j)
        acc[i][j] = wmma_f16(a[i].v, b[j].v, acc[i][j]);
    __syncthreads();
  }

  // Epilogue: which-matrix / head / batch are uniform per 16x16 D-tile.
  for (int j = 0; j < 2; ++j) {
    const int gnb   = n0 + wn * 32 + j * 16;      // tile-uniform column base
    const int which = gnb >> 10;                  // 0=Q 1=K 2=V
    const int hh    = (gnb & 1023) >> 6;
    const int dd    = (gnb & 63) + l16;           // this lane's d
    const float bv  = bias[gnb + l16];
    for (int i = 0; i < 2; ++i) {
      const int gmb = m0 + wm * 32 + i * 16;      // 16-row tile within one batch
      const int bb  = gmb >> 11;
      const size_t hb = (size_t)(bb * NHEAD + hh);
      for (int r = 0; r < 8; ++r) {
        int ll  = (gmb & 2047) + r + 8 * half;
        float v = acc[i][j][r] + bv;
        if (which == 0)      Qb[(hb * LL + ll) * HD + dd] = (_Float16)(v * 0.125f);
        else if (which == 1) Kb[(hb * LL + ll) * HD + dd] = (_Float16)v;
        else                 Vt[(hb * HD + dd) * LL + ll] = (_Float16)v;
      }
    }
  }
}

// ---------------------------------------------------------------------------
// Kernel 2: causal attention. Each wave owns 16 query rows.
// Pass 1: online softmax stats via streaming QK^T WMMA.
// Pass 2: recompute scores, write normalized probs, accumulate P.V via WMMA.
// ---------------------------------------------------------------------------
__global__ __launch_bounds__(256)
void attn_kernel(const _Float16* __restrict__ Qb, const _Float16* __restrict__ Kb,
                 const _Float16* __restrict__ Vt,
                 float* __restrict__ probs, _Float16* __restrict__ Ctx)
{
  __shared__ _Float16 Plds[8][16][40];      // per-wave 16x32 P staging

  const int t    = threadIdx.x;
  const int lane = t & 31;
  const int wid  = t >> 5;
  const int half = lane >> 4;
  const int l16  = lane & 15;
  const int hh   = blockIdx.y;
  const int bb   = blockIdx.z;
  const int q0   = blockIdx.x * 128 + wid * 16;

  const size_t bh = (size_t)(bb * NHEAD + hh);
  const _Float16* Qp = Qb + bh * LL * HD;
  const _Float16* Kp = Kb + bh * LL * HD;
  const _Float16* Vp = Vt + bh * HD * LL;
  float* Pr = probs + bh * (size_t)LL * LL;

  // Q A-fragments (2 chunks of K=32 over d), already scaled by 1/sqrt(D)
  AFrag16 qa[2];
  {
    const _Float16* qrow = Qp + (size_t)(q0 + l16) * HD;
    for (int c = 0; c < 2; ++c) {
      const _Float16* p = qrow + c * 32 + half * 8;
      qa[c].q[0] = *(const uint4*)p;
      qa[c].q[1] = *(const uint4*)(p + 16);
    }
  }

  float mrow[8], lrow[8];
  for (int r = 0; r < 8; ++r) { mrow[r] = -INFINITY; lrow[r] = 0.f; }

  const int last_t = q0 >> 4;   // last 16-wide k-tile touching the causal band

  // ---- Pass 1: running (max, sumexp) per row --------------------------------
  for (int kt = 0; kt <= last_t; ++kt) {
    int k0 = kt * 16;
    AFrag16 bk[2];
    const _Float16* krow = Kp + (size_t)(k0 + l16) * HD;
    for (int c = 0; c < 2; ++c) {
      const _Float16* p = krow + c * 32 + half * 8;
      bk[c].q[0] = *(const uint4*)p;
      bk[c].q[1] = *(const uint4*)(p + 16);
    }
    v8f s = {};
    s = wmma_f16(qa[0].v, bk[0].v, s);
    s = wmma_f16(qa[1].v, bk[1].v, s);
    int kcol = k0 + l16;
    for (int r = 0; r < 8; ++r) {
      int qg = q0 + r + 8 * half;
      float sv = (kcol <= qg) ? s[r] : -1.0e9f;
      float mn = fmaxf(mrow[r], sv);
      lrow[r]  = lrow[r] * __expf(mrow[r] - mn) + __expf(sv - mn);
      mrow[r]  = mn;
    }
  }
  // merge stats across the 16 lanes holding each row
  for (int r = 0; r < 8; ++r) {
    float m = mrow[r], l = lrow[r];
    for (int off = 1; off < 16; off <<= 1) {
      float mo = __shfl_xor(m, off, 32);
      float lo = __shfl_xor(l, off, 32);
      float mn = fmaxf(m, mo);
      l = l * __expf(m - mn) + lo * __expf(mo - mn);
      m = mn;
    }
    mrow[r] = m;
    lrow[r] = 1.0f / l;   // keep inverse
  }

  // ---- Pass 2: probs + context ---------------------------------------------
  v8f ctx[4];
  for (int dc = 0; dc < 4; ++dc) ctx[dc] = {};

  for (int kb = 0; kb < LL / 32; ++kb) {
    int k0 = kb * 32;
    if (k0 <= q0 + 15) {
      v8f ps[2];
      for (int t2 = 0; t2 < 2; ++t2) {
        int kt0 = k0 + t2 * 16;
        AFrag16 bk[2];
        const _Float16* krow = Kp + (size_t)(kt0 + l16) * HD;
        for (int c = 0; c < 2; ++c) {
          const _Float16* p = krow + c * 32 + half * 8;
          bk[c].q[0] = *(const uint4*)p;
          bk[c].q[1] = *(const uint4*)(p + 16);
        }
        v8f s = {};
        s = wmma_f16(qa[0].v, bk[0].v, s);
        s = wmma_f16(qa[1].v, bk[1].v, s);
        int kcol = kt0 + l16;
        for (int r = 0; r < 8; ++r) {
          int qg = q0 + r + 8 * half;
          float sv = (kcol <= qg) ? s[r] : -1.0e9f;
          s[r] = __expf(sv - mrow[r]) * lrow[r];
        }
        ps[t2] = s;
      }
      // write probs (f32) and stage P tile (f16) for the P.V WMMA
      for (int t2 = 0; t2 < 2; ++t2)
        for (int r = 0; r < 8; ++r) {
          int qg = q0 + r + 8 * half;
          Pr[(size_t)qg * LL + k0 + t2 * 16 + l16] = ps[t2][r];
          Plds[wid][r + 8 * half][t2 * 16 + l16] = (_Float16)ps[t2][r];
        }
      __asm__ volatile("s_wait_dscnt 0" ::: "memory");
      AFrag16 pa;
      {
        const _Float16* p = &Plds[wid][l16][half * 8];
        pa.q[0] = *(const uint4*)p;
        pa.q[1] = *(const uint4*)(p + 16);
      }
      for (int dc = 0; dc < 4; ++dc) {
        AFrag16 bv;
        const _Float16* p = Vp + (size_t)(dc * 16 + l16) * LL + k0 + half * 8;
        bv.q[0] = *(const uint4*)p;
        bv.q[1] = *(const uint4*)(p + 16);
        ctx[dc] = wmma_f16(pa.v, bv.v, ctx[dc]);
      }
    } else {
      // fully above the diagonal: probs are exactly zero -> float4 fill
      const float4 z4 = make_float4(0.f, 0.f, 0.f, 0.f);
      for (int i = 0; i < 4; ++i) {
        int id  = lane + 32 * i;            // 0..127 (16 rows x 8 float4)
        int row = id >> 3;
        int c4  = (id & 7) << 2;
        *(float4*)(&Pr[(size_t)(q0 + row) * LL + k0 + c4]) = z4;
      }
    }
  }

  // context -> [b, l, h*64+d] f16 for the output projection
  for (int dc = 0; dc < 4; ++dc)
    for (int r = 0; r < 8; ++r) {
      int qg  = q0 + r + 8 * half;
      int col = hh * HD + dc * 16 + l16;
      Ctx[((size_t)bb * LL + qg) * EMBED + col] = (_Float16)ctx[dc][r];
    }
}

// ---------------------------------------------------------------------------
// Kernel 3: out = ctx @ w_out + b_out  (M=8192, N=1024, K=1024)
// ---------------------------------------------------------------------------
__global__ __launch_bounds__(256)
void out_kernel(const _Float16* __restrict__ Ctx, const float* __restrict__ w,
                const float* __restrict__ bias, float* __restrict__ out)
{
  const int N = EMBED, K = EMBED;
  __shared__ _Float16 As[128][40];
  __shared__ _Float16 Bs[64][40];

  const int t    = threadIdx.x;
  const int lane = t & 31;
  const int wid  = t >> 5;
  const int wm   = wid >> 1;
  const int wn   = wid & 1;
  const int half = lane >> 4;
  const int l16  = lane & 15;
  const int m0   = blockIdx.x * 128;
  const int n0   = blockIdx.y * 64;

  v8f acc[2][2];
  for (int i = 0; i < 2; ++i) for (int j = 0; j < 2; ++j) acc[i][j] = {};

  for (int k0 = 0; k0 < K; k0 += 32) {
    for (int it = 0; it < 2; ++it) {
      int idx = t + it * 256;            // 0..511 (128 rows x 4 8-half groups)
      int row = idx >> 2;
      int g8  = (idx & 3) * 8;
      *(uint4*)&As[row][g8] = *(const uint4*)(Ctx + (size_t)(m0 + row) * K + k0 + g8);
    }
    for (int it = 0; it < 2; ++it) {
      int idx = t + it * 256;
      int kr  = idx >> 4;
      int c4  = (idx & 15) << 2;
      float4 f = *(const float4*)(w + (size_t)(k0 + kr) * N + n0 + c4);
      Bs[c4 + 0][kr] = (_Float16)f.x;  Bs[c4 + 1][kr] = (_Float16)f.y;
      Bs[c4 + 2][kr] = (_Float16)f.z;  Bs[c4 + 3][kr] = (_Float16)f.w;
    }
    __syncthreads();

    AFrag16 a[2], b[2];
    for (int i = 0; i < 2; ++i) {
      const _Float16* p = &As[wm * 32 + i * 16 + l16][half * 8];
      a[i].q[0] = *(const uint4*)p;
      a[i].q[1] = *(const uint4*)(p + 16);
    }
    for (int j = 0; j < 2; ++j) {
      const _Float16* p = &Bs[wn * 32 + j * 16 + l16][half * 8];
      b[j].q[0] = *(const uint4*)p;
      b[j].q[1] = *(const uint4*)(p + 16);
    }
    for (int i = 0; i < 2; ++i)
      for (int j = 0; j < 2; ++j)
        acc[i][j] = wmma_f16(a[i].v, b[j].v, acc[i][j]);
    __syncthreads();
  }

  for (int j = 0; j < 2; ++j) {
    const int gn = n0 + wn * 32 + j * 16 + l16;
    const float bv = bias[gn];
    for (int i = 0; i < 2; ++i)
      for (int r = 0; r < 8; ++r) {
        int gm = m0 + wm * 32 + i * 16 + r + 8 * half;
        out[(size_t)gm * N + gn] = acc[i][j][r] + bv;
      }
  }
}

// ---------------------------------------------------------------------------
extern "C" void kernel_launch(void* const* d_in, const int* in_sizes, int n_in,
                              void* d_out, int out_size, void* d_ws, size_t ws_size,
                              hipStream_t stream) {
  (void)in_sizes; (void)n_in; (void)out_size; (void)ws_size;
  const float* x     = (const float*)d_in[0];
  const float* w_qkv = (const float*)d_in[1];
  const float* b_qkv = (const float*)d_in[2];
  const float* w_out = (const float*)d_in[3];
  const float* b_out = (const float*)d_in[4];

  float* out   = (float*)d_out;                          // (B,L,E)
  float* probs = out + (size_t)BB * LL * EMBED;          // (B,H,L,L)

  const size_t elems = (size_t)BB * NHEAD * LL * HD;     // 8,388,608
  _Float16* Qb  = (_Float16*)d_ws;                       // 16 MiB each
  _Float16* Kb  = Qb + elems;
  _Float16* Vt  = Kb + elems;
  _Float16* Ctx = Vt + elems;                            // 64 MiB total

  qkv_kernel<<<dim3((BB * LL) / 128, (3 * EMBED) / 64), 256, 0, stream>>>(
      x, w_qkv, b_qkv, Qb, Kb, Vt);
  attn_kernel<<<dim3(LL / 128, NHEAD, BB), 256, 0, stream>>>(
      Qb, Kb, Vt, probs, Ctx);
  out_kernel<<<dim3((BB * LL) / 128, EMBED / 64), 256, 0, stream>>>(
      Ctx, w_out, b_out, out);
}
